// MCCEDecentralizedPolicyNetwork_56856777064457
// MI455X (gfx1250) — matrix operand order repaired and checked by reading, not compiled, not run on hardware
//
#include <hip/hip_runtime.h>

// ---------------- problem constants ----------------
#define A_DIM   8
#define BS_DIM  32
#define T_DIM   256
#define V_DIM   128
#define E_DIM   512
#define H_DIM   512
#define NA_DIM  64
#define NB      (A_DIM * BS_DIM)   // 256 sequences
#define MROWS   (NB * T_DIM)       // 65536 tokens
#define G_DIM   (3 * H_DIM)        // 1536

typedef __bf16 v16bf __attribute__((ext_vector_type(16)));
typedef float  v8f   __attribute__((ext_vector_type(8)));
typedef unsigned short u16;

#define WMMA_BF16(a, b, c) \
    __builtin_amdgcn_wmma_f32_16x16x32_bf16(false, (a), false, (b), (short)0, (c), false, false)

// ---------------- helpers ----------------
__device__ __forceinline__ u16 f2bf(float f) {
    unsigned int u = __float_as_uint(f);
    u += 0x7FFFu + ((u >> 16) & 1u);      // round-to-nearest-even
    return (u16)(u >> 16);
}
__device__ __forceinline__ float bf2f(u16 h) {
    return __uint_as_float(((unsigned int)h) << 16);
}

// A-fragment: 16x32 bf16 (MxK), row-major source, leading dim lda (elements).
// lane l holds row m = l&15; lanes<16 take K[0..7],K[16..23]; lanes>=16 take K[8..15],K[24..31].
__device__ __forceinline__ v16bf load_a_frag(const u16* A, int lda, int m0, int k0, int lane) {
    const int m    = m0 + (lane & 15);
    const int half = lane >> 4;
    const u16* p = A + (size_t)m * lda + k0 + half * 8;
    v16bf a;
    ((uint4*)&a)[0] = *(const uint4*)p;          // K chunk: base
    ((uint4*)&a)[1] = *(const uint4*)(p + 16);   // K chunk: base+16
    return a;
}

// B-fragment: 32x16 bf16 (KxN). Weights stored as W[n, k] (K contiguous per output col).
// lane l holds col n = l&15; lanes<16 take K[0..7],K[16..23]; lanes>=16 take K[8..15],K[24..31].
__device__ __forceinline__ v16bf load_b_frag(const u16* W, int ldw, int n0, int k0, int lane) {
    const int n  = n0 + (lane & 15);
    const int kb = k0 + (lane >> 4) * 8;
    const u16* p = W + (size_t)n * ldw + kb;
    v16bf b;
    ((uint4*)&b)[0] = *(const uint4*)p;
    ((uint4*)&b)[1] = *(const uint4*)(p + 16);
    return b;
}

// ---------------- fp32 -> bf16 convert ----------------
__global__ void cvt_f32_bf16_kernel(const float* __restrict__ in, u16* __restrict__ out, int n) {
    int i = blockIdx.x * blockDim.x + threadIdx.x;
    if (i < n) out[i] = f2bf(in[i]);
}

// ---------------- generic bf16 WMMA GEMM: out = act(A * W^T + bias) ----------------
// A: [M,K] bf16, W: [N,K] bf16, out bf16 (and/or fp32) [M,N].
// One wave -> 32x64 tile: 2 A frags x 4 B frags -> 8 accumulators (8 WMMAs per 6 fragment
// loads), register double-buffered across K so loads stay ahead of the matrix pipe.
template <int K>
__global__ void gemm_bias_kernel(const u16* __restrict__ A, const u16* __restrict__ W,
                                 const float* __restrict__ bias,
                                 u16* __restrict__ outBf, float* __restrict__ outF32,
                                 int N, int relu) {
    const int lane = threadIdx.x & 31;
    const int wid  = threadIdx.x >> 5;
    const int ctiles = N >> 6;                       // 64-wide col tiles
    const int tile = blockIdx.x * 8 + wid;
    const int rt = tile / ctiles, ct = tile % ctiles;
    const int m0 = rt * 32, n0 = ct * 64;

    v8f acc[2][4] = {};
    v16bf a_cur[2], b_cur[4];
#pragma unroll
    for (int i = 0; i < 2; ++i) a_cur[i] = load_a_frag(A, K, m0 + i * 16, 0, lane);
#pragma unroll
    for (int j = 0; j < 4; ++j) b_cur[j] = load_b_frag(W, K, n0 + j * 16, 0, lane);

#pragma unroll
    for (int k0 = 32; k0 < K; k0 += 32) {
        v16bf a_nxt[2], b_nxt[4];
#pragma unroll
        for (int i = 0; i < 2; ++i) a_nxt[i] = load_a_frag(A, K, m0 + i * 16, k0, lane);
#pragma unroll
        for (int j = 0; j < 4; ++j) b_nxt[j] = load_b_frag(W, K, n0 + j * 16, k0, lane);
#pragma unroll
        for (int i = 0; i < 2; ++i)
#pragma unroll
            for (int j = 0; j < 4; ++j) acc[i][j] = WMMA_BF16(a_cur[i], b_cur[j], acc[i][j]);
#pragma unroll
        for (int i = 0; i < 2; ++i) a_cur[i] = a_nxt[i];
#pragma unroll
        for (int j = 0; j < 4; ++j) b_cur[j] = b_nxt[j];
    }
#pragma unroll
    for (int i = 0; i < 2; ++i)
#pragma unroll
        for (int j = 0; j < 4; ++j) acc[i][j] = WMMA_BF16(a_cur[i], b_cur[j], acc[i][j]);

    const int n = lane & 15, half = lane >> 4;
#pragma unroll
    for (int i = 0; i < 2; ++i) {
#pragma unroll
        for (int j = 0; j < 4; ++j) {
            const int col = n0 + j * 16 + n;
            const float bv = bias[col];
#pragma unroll
            for (int v = 0; v < 8; ++v) {
                const int row = m0 + i * 16 + v + 8 * half;
                float x = acc[i][j][v] + bv;
                if (relu) x = fmaxf(x, 0.0f);
                const size_t idx = (size_t)row * N + col;
                if (outBf)  outBf[idx]  = f2bf(x);
                if (outF32) outF32[idx] = x;
            }
        }
    }
}

// ---------------- one GRU time step ----------------
// gh = h_{t-1} * Whh^T (3 gates) + fused gate math over the full [NB,H] state.
// Latency-bound: spread as 256 single-wave workgroups; each wave owns a 16x32 tile
// (6 accumulators, double-buffered A fragment).
__global__ void gru_step_kernel(const u16* __restrict__ Aprev, int lda,       // bf16 h_{t-1}
                                const float* __restrict__ Hprev, int ldh,     // fp32 h_{t-1}
                                const u16* __restrict__ Whh,                  // [3H,H] bf16
                                const float* __restrict__ b_hh,
                                const u16* __restrict__ gi,                   // bf16 [MROWS,3H]
                                u16* __restrict__ h_bf_all,                   // bf16 [MROWS,H]
                                float* __restrict__ h_out,                    // fp32 [MROWS,H]
                                int t) {
    const int lane = threadIdx.x & 31;
    const int ctiles = H_DIM >> 5;                   // 16 col tiles of 32
    const int tile = blockIdx.x;                     // 0..255
    const int rt = tile / ctiles, ct = tile % ctiles;
    const int m0 = rt * 16, c0 = ct * 32;

    v8f aR[2] = {v8f{0}, v8f{0}};
    v8f aZ[2] = {v8f{0}, v8f{0}};
    v8f aN[2] = {v8f{0}, v8f{0}};
    v16bf a_cur = load_a_frag(Aprev, lda, m0, 0, lane);
#pragma unroll
    for (int k0 = 0; k0 < H_DIM; k0 += 32) {
        // prefetch B fragments for this K-block and A for the next one
        v16bf bR[2], bZ[2], bN[2];
#pragma unroll
        for (int j = 0; j < 2; ++j) {
            const int cn = c0 + j * 16;
            bR[j] = load_b_frag(Whh, H_DIM, cn,             k0, lane);
            bZ[j] = load_b_frag(Whh, H_DIM, H_DIM + cn,     k0, lane);
            bN[j] = load_b_frag(Whh, H_DIM, 2 * H_DIM + cn, k0, lane);
        }
        v16bf a_nxt;
        if (k0 + 32 < H_DIM) a_nxt = load_a_frag(Aprev, lda, m0, k0 + 32, lane);
#pragma unroll
        for (int j = 0; j < 2; ++j) {
            aR[j] = WMMA_BF16(a_cur, bR[j], aR[j]);
            aZ[j] = WMMA_BF16(a_cur, bZ[j], aZ[j]);
            aN[j] = WMMA_BF16(a_cur, bN[j], aN[j]);
        }
        a_cur = a_nxt;
    }
    const int n = lane & 15, half = lane >> 4;
#pragma unroll
    for (int j = 0; j < 2; ++j) {
        const int col = c0 + j * 16 + n;
        const float br = b_hh[col];
        const float bz = b_hh[H_DIM + col];
        const float bn = b_hh[2 * H_DIM + col];
#pragma unroll
        for (int v = 0; v < 8; ++v) {
            const int row = m0 + v + 8 * half;                    // sequence index
            const u16* gp = gi + ((size_t)row * T_DIM + t) * G_DIM;
            const float ir  = bf2f(gp[col]);
            const float iz  = bf2f(gp[H_DIM + col]);
            const float in_ = bf2f(gp[2 * H_DIM + col]);
            const float r  = 1.0f / (1.0f + __expf(-(ir + aR[j][v] + br)));
            const float z  = 1.0f / (1.0f + __expf(-(iz + aZ[j][v] + bz)));
            const float nn = tanhf(in_ + r * (aN[j][v] + bn));
            const float hp = Hprev[(size_t)row * ldh + col];
            const float hn = (1.0f - z) * nn + z * hp;
            const size_t oidx = ((size_t)row * T_DIM + t) * H_DIM + col;
            h_out[oidx]    = hn;
            h_bf_all[oidx] = f2bf(hn);
        }
    }
}

// ---------------- decoder GEMM + bias + ReLU + softmax over NA=64 ----------------
// One wave -> 32 rows x all 64 cols (2 A frags x 4 B frags -> 8 accumulators).
// Row softmax via width-16 shuffles per half-wave.
__global__ void decoder_softmax_kernel(const u16* __restrict__ Hbf,           // [MROWS,H] bf16
                                       const u16* __restrict__ DecW,          // [NA,H] bf16
                                       const float* __restrict__ dec_b,
                                       float* __restrict__ pi) {
    const int lane = threadIdx.x & 31;
    const int wid  = threadIdx.x >> 5;
    const int m0 = (blockIdx.x * 8 + wid) * 32;

    v8f acc[2][4] = {};
    v16bf a_cur[2], b_cur[4];
#pragma unroll
    for (int i = 0; i < 2; ++i) a_cur[i] = load_a_frag(Hbf, H_DIM, m0 + i * 16, 0, lane);
#pragma unroll
    for (int j = 0; j < 4; ++j) b_cur[j] = load_b_frag(DecW, H_DIM, j * 16, 0, lane);
#pragma unroll
    for (int k0 = 32; k0 < H_DIM; k0 += 32) {
        v16bf a_nxt[2], b_nxt[4];
#pragma unroll
        for (int i = 0; i < 2; ++i) a_nxt[i] = load_a_frag(Hbf, H_DIM, m0 + i * 16, k0, lane);
#pragma unroll
        for (int j = 0; j < 4; ++j) b_nxt[j] = load_b_frag(DecW, H_DIM, j * 16, k0, lane);
#pragma unroll
        for (int i = 0; i < 2; ++i)
#pragma unroll
            for (int j = 0; j < 4; ++j) acc[i][j] = WMMA_BF16(a_cur[i], b_cur[j], acc[i][j]);
#pragma unroll
        for (int i = 0; i < 2; ++i) a_cur[i] = a_nxt[i];
#pragma unroll
        for (int j = 0; j < 4; ++j) b_cur[j] = b_nxt[j];
    }
#pragma unroll
    for (int i = 0; i < 2; ++i)
#pragma unroll
        for (int j = 0; j < 4; ++j) acc[i][j] = WMMA_BF16(a_cur[i], b_cur[j], acc[i][j]);

    const int n = lane & 15, half = lane >> 4;
#pragma unroll
    for (int i = 0; i < 2; ++i) {
#pragma unroll
        for (int v = 0; v < 8; ++v) {
            float vals[4];
#pragma unroll
            for (int j = 0; j < 4; ++j)
                vals[j] = fmaxf(acc[i][j][v] + dec_b[j * 16 + n], 0.0f);
            float mx = fmaxf(fmaxf(vals[0], vals[1]), fmaxf(vals[2], vals[3]));
            mx = fmaxf(mx, __shfl_xor(mx, 8, 16));
            mx = fmaxf(mx, __shfl_xor(mx, 4, 16));
            mx = fmaxf(mx, __shfl_xor(mx, 2, 16));
            mx = fmaxf(mx, __shfl_xor(mx, 1, 16));
            float e[4], s = 0.0f;
#pragma unroll
            for (int j = 0; j < 4; ++j) { e[j] = __expf(vals[j] - mx); s += e[j]; }
            s += __shfl_xor(s, 8, 16);
            s += __shfl_xor(s, 4, 16);
            s += __shfl_xor(s, 2, 16);
            s += __shfl_xor(s, 1, 16);
            const float inv = 1.0f / s;
            const int row = m0 + i * 16 + v + 8 * half;
#pragma unroll
            for (int j = 0; j < 4; ++j)
                pi[(size_t)row * NA_DIM + j * 16 + n] = e[j] * inv;
        }
    }
}

// ---------------- launcher ----------------
extern "C" void kernel_launch(void* const* d_in, const int* in_sizes, int n_in,
                              void* d_out, int out_size, void* d_ws, size_t ws_size,
                              hipStream_t stream) {
    (void)in_sizes; (void)n_in; (void)out_size; (void)ws_size;
    const float* x      = (const float*)d_in[0];  // [8,32,256,128]
    const float* h0f    = (const float*)d_in[1];  // [8,32,1,512] == [256,512]
    const float* enc_w  = (const float*)d_in[2];  // [512,128]
    const float* enc_b  = (const float*)d_in[3];  // [512]
    const float* w_ih   = (const float*)d_in[4];  // [1536,512]
    const float* w_hh   = (const float*)d_in[5];  // [1536,512]
    const float* b_ih   = (const float*)d_in[6];  // [1536]
    const float* b_hh   = (const float*)d_in[7];  // [1536]
    const float* dec_w  = (const float*)d_in[8];  // [64,512]
    const float* dec_b  = (const float*)d_in[9];  // [64]

    float* pi_out = (float*)d_out;                       // [MROWS, 64]
    float* h_out  = pi_out + (size_t)MROWS * NA_DIM;     // [MROWS, 512] (rows = n*T + t)

    // workspace carve-up (all sizes 256B-aligned)
    char* w = (char*)d_ws;
    u16* x_bf    = (u16*)w; w += (size_t)MROWS * V_DIM * 2;        // 16 MB
    u16* encw_bf = (u16*)w; w += (size_t)E_DIM * V_DIM * 2;        // 128 KB
    u16* wih_bf  = (u16*)w; w += (size_t)G_DIM * E_DIM * 2;        // 1.5 MB
    u16* whh_bf  = (u16*)w; w += (size_t)G_DIM * H_DIM * 2;        // 1.5 MB
    u16* decw_bf = (u16*)w; w += (size_t)NA_DIM * H_DIM * 2;       // 64 KB
    u16* h0_bf   = (u16*)w; w += (size_t)NB * H_DIM * 2;           // 256 KB
    u16* enc_bf  = (u16*)w; w += (size_t)MROWS * E_DIM * 2;        // 64 MB
    u16* h_bf    = (u16*)w; w += (size_t)MROWS * H_DIM * 2;        // 64 MB
    u16* gi_bf   = (u16*)w; w += (size_t)MROWS * G_DIM * 2;        // 192 MB

    // 1) fp32 -> bf16 conversions
    auto cvt = [&](const float* in, u16* out, int nelem) {
        cvt_f32_bf16_kernel<<<(nelem + 255) / 256, 256, 0, stream>>>(in, out, nelem);
    };
    cvt(x,     x_bf,    MROWS * V_DIM);
    cvt(enc_w, encw_bf, E_DIM * V_DIM);
    cvt(w_ih,  wih_bf,  G_DIM * E_DIM);
    cvt(w_hh,  whh_bf,  G_DIM * H_DIM);
    cvt(dec_w, decw_bf, NA_DIM * H_DIM);
    cvt(h0f,   h0_bf,   NB * H_DIM);

    // 2) encoder: enc = relu(x @ enc_w^T + enc_b)   [MROWS, E]
    {
        int tiles = (MROWS / 32) * (E_DIM / 64);
        gemm_bias_kernel<V_DIM><<<tiles / 8, 256, 0, stream>>>(
            x_bf, encw_bf, enc_b, enc_bf, nullptr, E_DIM, 1);
    }
    // 3) input projections (hoisted out of the recurrence): gi = enc @ w_ih^T + b_ih  [MROWS, 3H]
    {
        int tiles = (MROWS / 32) * (G_DIM / 64);
        gemm_bias_kernel<E_DIM><<<tiles / 8, 256, 0, stream>>>(
            enc_bf, wih_bf, b_ih, gi_bf, nullptr, G_DIM, 0);
    }
    // 4) GRU recurrence: one launch per step (graph replay amortizes launch cost)
    for (int t = 0; t < T_DIM; ++t) {
        const u16*   aprev; int lda;
        const float* hprev; int ldh;
        if (t == 0) { aprev = h0_bf;                       lda = H_DIM;
                      hprev = h0f;                          ldh = H_DIM; }
        else        { aprev = h_bf  + (size_t)(t - 1) * H_DIM; lda = T_DIM * H_DIM;
                      hprev = h_out + (size_t)(t - 1) * H_DIM; ldh = T_DIM * H_DIM; }
        gru_step_kernel<<<256, 32, 0, stream>>>(aprev, lda, hprev, ldh,
                                                whh_bf, b_hh, gi_bf, h_bf, h_out, t);
    }
    // 5) decoder + softmax: pi = softmax(relu(h @ dec_w^T + dec_b))
    {
        int tiles = MROWS / 32;
        decoder_softmax_kernel<<<tiles / 8, 256, 0, stream>>>(h_bf, decw_bf, dec_b, pi_out);
    }
}
